// Pose_Module_72018011619457
// MI455X (gfx1250) — matrix operand order repaired and checked by reading, not compiled, not run on hardware
//
#include <hip/hip_runtime.h>
#include <math.h>

// ---------------------------------------------------------------------------
// Types for WMMA fragments (wave32, CDNA5 16x16x32 f16 -> f32)
// ---------------------------------------------------------------------------
typedef _Float16 h8  __attribute__((ext_vector_type(8)));
typedef _Float16 h16 __attribute__((ext_vector_type(16)));
typedef float    f8  __attribute__((ext_vector_type(8)));

// Model dims
#define BB   16
#define CC   32
#define TT   62
#define VV   19
#define DD   128
#define HD   256
#define NH   8
#define HDIM 16
#define NTOK 18848           // B*T*V = B*V*T
#define NPAIR 358112         // B*T*V*V

// Async copy helpers (CDNA5 gfx1250): global -> LDS, tracked by ASYNCcnt.
__device__ __forceinline__ void async_b128(unsigned lds_off, const void* gptr) {
    unsigned long long ga = (unsigned long long)gptr;
    asm volatile("global_load_async_to_lds_b128 %0, %1, off"
                 :: "v"(lds_off), "v"(ga) : "memory");
}
__device__ __forceinline__ unsigned lds_off(const void* p) {
    return (unsigned)(unsigned long long)p;   // low 32 bits of flat addr = LDS offset
}

// ---------------------------------------------------------------------------
// Kernel 0: transpose pose [B,C,T,V] -> P [B*T*V, C] (f32)
// ---------------------------------------------------------------------------
__global__ void k_transpose(const float* __restrict__ pose, float* __restrict__ P) {
    int idx = blockIdx.x * blockDim.x + threadIdx.x;
    if (idx >= NTOK * CC) return;
    int c   = idx & 31;
    int row = idx >> 5;          // (b*T+t)*V + v
    int v   = row % VV;
    int bt  = row / VV;
    int t   = bt % TT;
    int b   = bt / TT;
    P[idx] = pose[((b * CC + c) * TT + t) * VV + v];
}

// ---------------------------------------------------------------------------
// Kernel 1: sinusoidal positional encoding pe[62][128]
// ---------------------------------------------------------------------------
__global__ void k_pe(float* __restrict__ pe) {
    int idx = blockIdx.x * blockDim.x + threadIdx.x;
    if (idx >= TT * DD) return;
    int t = idx / DD;
    int d = idx % DD;
    int p = d >> 1;
    float freq = __expf((float)(2 * p) * (-9.210340371976184f / (float)DD));
    float ang  = (float)t * freq;
    pe[idx] = (d & 1) ? cosf(ang) : sinf(ang);
}

// ---------------------------------------------------------------------------
// Generic WMMA GEMM: out[M,N] = act( A[M,K] @ W[N,K]^T + bias[N] (+ pe) )
// Block: 256 threads = 8 waves; block tile 32(M) x 64(N); K step 32.
// Double-buffered async global->LDS staging (ASYNCcnt), f32 tiles in LDS,
// f32->f16 conversion fused into the fragment gather (co-executes with WMMA).
// Requires M%32==0, N%64==0, K%32==0 (all GEMMs here satisfy this).
// ACT: 0 none, 1 relu, 2 leakyrelu(0.01)
// ---------------------------------------------------------------------------
#define ASTF 36   // LDS row stride (floats) for A tile (32 rows x 32 k)
#define BSTF 36   // LDS row stride (floats) for B tile (64 rows x 32 k)

template<int ACT, bool HAS_BIAS, bool HAS_PE>
__global__ __launch_bounds__(256) void k_gemm(
    const float* __restrict__ A, const float* __restrict__ W,
    const float* __restrict__ bias, const float* __restrict__ pe,
    float* __restrict__ out, int M, int N, int K)
{
    __shared__ __align__(16) float Af[2][32 * ASTF];
    __shared__ __align__(16) float Bf[2][64 * BSTF];

    const int tid  = threadIdx.x;
    const int m0   = blockIdx.x * 32;
    const int n0   = blockIdx.y * 64;
    const int wave = tid >> 5;
    const int lane = tid & 31;
    const int wm   = (wave & 1) * 16;   // wave M offset in tile
    const int wn   = (wave >> 1) * 16;  // wave N offset in tile
    const int lg   = lane >> 4;         // lane group (0/1)
    const int l15  = lane & 15;

    // per-thread staging assignment
    const int ar = tid >> 3, ac = (tid & 7) * 4;   // A: 1 x b128 per thread
    const int br = tid >> 2, bc = (tid & 3) * 8;   // B: 2 x b128 per thread
    const float* gA = A + (size_t)(m0 + ar) * K + ac;
    const float* gB = W + (size_t)(n0 + br) * K + bc;
    const unsigned la0 = lds_off(&Af[0][ar * ASTF + ac]);
    const unsigned la1 = lds_off(&Af[1][ar * ASTF + ac]);
    const unsigned lb0 = lds_off(&Bf[0][br * BSTF + bc]);
    const unsigned lb1 = lds_off(&Bf[1][br * BSTF + bc]);

    f8 acc = {0.f, 0.f, 0.f, 0.f, 0.f, 0.f, 0.f, 0.f};

    // prologue: issue tile 0 into buffer 0
    async_b128(la0, gA);
    async_b128(lb0, gB);
    async_b128(lb0 + 16u, gB + 4);

    const int nk = K >> 5;
    for (int it = 0; it < nk; ++it) {
        const int b = it & 1;
        if (it + 1 < nk) {
            // prefetch next tile into other buffer, then wait for current tile
            const int k1 = (it + 1) * 32;
            async_b128(b ? la0 : la1, gA + k1);
            async_b128(b ? lb0 : lb1, gB + k1);
            async_b128((b ? lb0 : lb1) + 16u, gB + k1 + 4);
            asm volatile("s_wait_asynccnt 0x3" ::: "memory");
        } else {
            asm volatile("s_wait_asynccnt 0x0" ::: "memory");
        }
        __syncthreads();   // current tile visible to all waves

        // ---- fragments (f32 LDS -> f16 regs) ----
        // A: lane l15 = row M, k = [8g..8g+7] and [16+8g..23+8g]
        const float* arow = &Af[b][(wm + l15) * ASTF + 8 * lg];
        const float4 a0 = *(const float4*)(arow);
        const float4 a1 = *(const float4*)(arow + 4);
        const float4 a2 = *(const float4*)(arow + 16);
        const float4 a3 = *(const float4*)(arow + 20);
        // B: lane l15 = col N, k = [8g..8g+7] and [16+8g..23+8g]
        const float* brow = &Bf[b][(wn + l15) * BSTF + 8 * lg];
        const float4 b0 = *(const float4*)(brow);
        const float4 b1 = *(const float4*)(brow + 4);
        const float4 b2 = *(const float4*)(brow + 16);
        const float4 b3 = *(const float4*)(brow + 20);

        h16 af, bf;
        af[0]  = (_Float16)a0.x; af[1]  = (_Float16)a0.y; af[2]  = (_Float16)a0.z; af[3]  = (_Float16)a0.w;
        af[4]  = (_Float16)a1.x; af[5]  = (_Float16)a1.y; af[6]  = (_Float16)a1.z; af[7]  = (_Float16)a1.w;
        af[8]  = (_Float16)a2.x; af[9]  = (_Float16)a2.y; af[10] = (_Float16)a2.z; af[11] = (_Float16)a2.w;
        af[12] = (_Float16)a3.x; af[13] = (_Float16)a3.y; af[14] = (_Float16)a3.z; af[15] = (_Float16)a3.w;
        bf[0]  = (_Float16)b0.x; bf[1]  = (_Float16)b0.y; bf[2]  = (_Float16)b0.z; bf[3]  = (_Float16)b0.w;
        bf[4]  = (_Float16)b1.x; bf[5]  = (_Float16)b1.y; bf[6]  = (_Float16)b1.z; bf[7]  = (_Float16)b1.w;
        bf[8]  = (_Float16)b2.x; bf[9]  = (_Float16)b2.y; bf[10] = (_Float16)b2.z; bf[11] = (_Float16)b2.w;
        bf[12] = (_Float16)b3.x; bf[13] = (_Float16)b3.y; bf[14] = (_Float16)b3.z; bf[15] = (_Float16)b3.w;

        acc = __builtin_amdgcn_wmma_f32_16x16x32_f16(
            false, af, false, bf, (short)0, acc, false, false);

        __syncthreads();   // all waves done reading buffer b before it is refilled
    }

    // ---- epilogue ----
    const int n = n0 + wn + l15;
    float bv = HAS_BIAS ? bias[n] : 0.f;
    #pragma unroll
    for (int r = 0; r < 8; r++) {
        int m = m0 + wm + r + 8 * lg;
        float val = acc[r] + bv;
        if (HAS_PE) val += pe[(m % TT) * DD + n];
        if (ACT == 1) val = fmaxf(val, 0.f);
        if (ACT == 2) val = (val >= 0.f) ? val : 0.01f * val;
        out[(size_t)m * N + n] = val;
    }
}

// ---------------------------------------------------------------------------
// Fused pairwise-bias MLP:
//   diff[16,32] = P[vi]-P[vj]  ->  h = diff @ Wemb^T (16x256, WMMA, PReLU)
//   bias8 = h @ Wbias^T * 1/sqrt(8)  (16x8, WMMA K=256)
// One wave per 16-row tile; h lives only in LDS (avoids 367 MB intermediate).
// ---------------------------------------------------------------------------
#define DST 32    // diff LDS stride (halves)
#define HST 264   // hidden LDS stride (halves) = 256 + 8 pad

__global__ __launch_bounds__(128) void k_biasmlp(
    const float* __restrict__ P, const float* __restrict__ Wemb,
    const float* __restrict__ prelua, const float* __restrict__ Wbias,
    float* __restrict__ outb, int ntiles)
{
    __shared__ __align__(16) _Float16 We[256 * 32];     // [n][k], n=hidden, k=channel
    __shared__ __align__(16) _Float16 Wb[16 * HST];     // [n padded to 16][k=256]
    __shared__ __align__(16) _Float16 diffs[4 * 16 * DST];
    __shared__ __align__(16) _Float16 hbuf[4 * 16 * HST];

    const int tid = threadIdx.x;  // 128 = 4 waves
    // stage W_emb (256x32) as f16
    for (int e = tid; e < 256 * 32; e += 128)
        We[e] = (_Float16)Wemb[e];
    // stage W_bias (8x256), pad n to 16 with zeros
    for (int e = tid; e < 16 * 256; e += 128) {
        int n = e >> 8, k = e & 255;
        Wb[n * HST + k] = (_Float16)((n < 8) ? Wbias[n * 256 + k] : 0.f);
    }
    __syncthreads();

    const float pa  = prelua[0];
    const int wave  = tid >> 5;
    const int lane  = tid & 31;
    const int tile  = blockIdx.x * 4 + wave;
    if (tile >= ntiles) return;   // no block barriers past this point

    _Float16* dptr = diffs + wave * 16 * DST;
    _Float16* hptr = hbuf  + wave * 16 * HST;

    // ---- pairwise diffs for this wave's 16 rows ----
    for (int e = lane; e < 512; e += 32) {
        int r = e >> 5, c = e & 31;
        int R   = tile * 16 + r;
        int bt  = R / 361;
        int rem = R - bt * 361;
        int vi  = rem / VV;
        int vj  = rem - vi * VV;
        float d = P[(bt * VV + vi) * CC + c] - P[(bt * VV + vj) * CC + c];
        dptr[r * DST + c] = (_Float16)d;
    }
    __builtin_amdgcn_wave_barrier();

    const int lg = lane >> 4, l15 = lane & 15;

    // ---- A fragment from diffs (single K=32 step) ----
    h8 alo = *(const h8*)(dptr + l15 * DST + 8 * lg);
    h8 ahi = *(const h8*)(dptr + l15 * DST + 16 + 8 * lg);
    h16 af;
    #pragma unroll
    for (int i = 0; i < 8; i++) { af[i] = alo[i]; af[i + 8] = ahi[i]; }

    // ---- GEMM1: 16 N-tiles of 16 -> hidden(256) with PReLU, stash in LDS ----
    for (int nt = 0; nt < 16; nt++) {
        h8 blo = *(const h8*)(We + (nt * 16 + l15) * 32 + 8 * lg);
        h8 bhi = *(const h8*)(We + (nt * 16 + l15) * 32 + 16 + 8 * lg);
        h16 bf;
        #pragma unroll
        for (int i = 0; i < 8; i++) { bf[i] = blo[i]; bf[i + 8] = bhi[i]; }
        f8 c = {0.f, 0.f, 0.f, 0.f, 0.f, 0.f, 0.f, 0.f};
        c = __builtin_amdgcn_wmma_f32_16x16x32_f16(
            false, af, false, bf, (short)0, c, false, false);
        #pragma unroll
        for (int r = 0; r < 8; r++) {
            float v = c[r];
            v = (v >= 0.f) ? v : pa * v;                 // PReLU
            hptr[(r + 8 * lg) * HST + nt * 16 + l15] = (_Float16)v;
        }
    }
    __builtin_amdgcn_wave_barrier();

    // ---- GEMM2: [16,256] x [256,8(pad16)] ----
    f8 c2 = {0.f, 0.f, 0.f, 0.f, 0.f, 0.f, 0.f, 0.f};
    for (int k0 = 0; k0 < 256; k0 += 32) {
        h8 a2lo = *(const h8*)(hptr + l15 * HST + k0 + 8 * lg);
        h8 a2hi = *(const h8*)(hptr + l15 * HST + k0 + 16 + 8 * lg);
        h8 b2lo = *(const h8*)(Wb + l15 * HST + k0 + 8 * lg);
        h8 b2hi = *(const h8*)(Wb + l15 * HST + k0 + 16 + 8 * lg);
        h16 a2, b2;
        #pragma unroll
        for (int i = 0; i < 8; i++) {
            a2[i] = a2lo[i]; a2[i + 8] = a2hi[i];
            b2[i] = b2lo[i]; b2[i + 8] = b2hi[i];
        }
        c2 = __builtin_amdgcn_wmma_f32_16x16x32_f16(
            false, a2, false, b2, (short)0, c2, false, false);
    }
    const float sc = 0.35355339059327373f;   // 1/sqrt(H)
    if (l15 < 8) {
        #pragma unroll
        for (int r = 0; r < 8; r++) {
            int R = tile * 16 + r + 8 * lg;
            outb[R * 8 + l15] = c2[r] * sc;
        }
    }
}

// ---------------------------------------------------------------------------
// Attention per (chunk, head): N rows, head_dim 16, optional additive bias,
// stable softmax. N=19 (graph, bias) or N=62 (temporal).
// q/k/v staged straight into LDS with async b128 copies (ASYNCcnt).
// ---------------------------------------------------------------------------
template<int N, bool HAS_BIAS>
__global__ __launch_bounds__(64) void k_attn(
    const float* __restrict__ qkv, const float* __restrict__ biasbuf,
    float* __restrict__ out)
{
    __shared__ __align__(16) float qs[N][HDIM];
    __shared__ __align__(16) float ks[N][HDIM];
    __shared__ __align__(16) float vs[N][HDIM];
    const int chunk = blockIdx.x;
    const int head  = blockIdx.y;
    const int tid   = threadIdx.x;

    for (int e = tid; e < N * 4; e += 64) {
        int j = e >> 2, c = (e & 3) * 4;
        const float* g = qkv + (size_t)(chunk * N + j) * (3 * DD) + head * HDIM + c;
        async_b128(lds_off(&qs[j][c]), g);
        async_b128(lds_off(&ks[j][c]), g + DD);
        async_b128(lds_off(&vs[j][c]), g + 2 * DD);
    }
    asm volatile("s_wait_asynccnt 0x0" ::: "memory");
    __syncthreads();

    const float scale = 0.25f;   // 1/sqrt(16)
    for (int i = tid; i < N; i += 64) {
        float s[N];
        float mx = -1e30f;
        for (int j = 0; j < N; j++) {
            float d = 0.f;
            #pragma unroll
            for (int dd = 0; dd < HDIM; dd++) d += qs[i][dd] * ks[j][dd];
            d *= scale;
            if (HAS_BIAS) d += biasbuf[((size_t)chunk * 361 + i * VV + j) * NH + head];
            s[j] = d;
            mx = fmaxf(mx, d);
        }
        float den = 0.f;
        for (int j = 0; j < N; j++) { float e = __expf(s[j] - mx); s[j] = e; den += e; }
        const float inv = 1.f / den;
        #pragma unroll
        for (int dd = 0; dd < HDIM; dd++) {
            float o = 0.f;
            for (int j = 0; j < N; j++) o += s[j] * vs[j][dd];
            out[(size_t)(chunk * N + i) * DD + head * HDIM + dd] = o * inv;
        }
    }
}

// ---------------------------------------------------------------------------
// LayerNorm: out = LN(x + r) * g + b (+ post).  One wave per row (D=128).
// ---------------------------------------------------------------------------
__global__ __launch_bounds__(256) void k_ln(
    const float* __restrict__ x, const float* __restrict__ r,
    const float* __restrict__ g, const float* __restrict__ b,
    const float* __restrict__ post, float* __restrict__ out, int nrows)
{
    const int row  = blockIdx.x * 8 + (threadIdx.x >> 5);
    const int lane = threadIdx.x & 31;
    if (row >= nrows) return;

    const float4 xv = *(const float4*)(x + (size_t)row * DD + lane * 4);
    const float4 rv = *(const float4*)(r + (size_t)row * DD + lane * 4);
    float v[4] = {xv.x + rv.x, xv.y + rv.y, xv.z + rv.z, xv.w + rv.w};

    float s = v[0] + v[1] + v[2] + v[3];
    for (int off = 16; off > 0; off >>= 1) s += __shfl_xor(s, off, 32);
    const float mean = s * (1.f / (float)DD);

    float q = 0.f;
    #pragma unroll
    for (int j = 0; j < 4; j++) { float d = v[j] - mean; q += d * d; }
    for (int off = 16; off > 0; off >>= 1) q += __shfl_xor(q, off, 32);
    const float rstd = rsqrtf(q * (1.f / (float)DD) + 1e-5f);

    #pragma unroll
    for (int j = 0; j < 4; j++) {
        int c = lane * 4 + j;
        float o = (v[j] - mean) * rstd * g[c] + b[c];
        if (post) o += post[(size_t)row * DD + c];
        out[(size_t)row * DD + c] = o;
    }
}

// ---------------------------------------------------------------------------
// Host orchestration
// ---------------------------------------------------------------------------
extern "C" void kernel_launch(void* const* d_in, const int* in_sizes, int n_in,
                              void* d_out, int out_size, void* d_ws, size_t ws_size,
                              hipStream_t stream) {
    (void)in_sizes; (void)n_in; (void)out_size; (void)ws_size;
    const float* pose    = (const float*)d_in[0];
    const float* W_pose  = (const float*)d_in[1];
    const float* W_emb   = (const float*)d_in[2];
    const float* prelu_a = (const float*)d_in[3];
    const float* W_bias  = (const float*)d_in[4];
    const float* gWqkv   = (const float*)d_in[5];
    const float* gbqkv   = (const float*)d_in[6];
    const float* gWo     = (const float*)d_in[7];
    const float* gbo     = (const float*)d_in[8];
    const float* gln1g   = (const float*)d_in[9];
    const float* gln1b   = (const float*)d_in[10];
    const float* gW1     = (const float*)d_in[11];
    const float* gb1     = (const float*)d_in[12];
    const float* gW2     = (const float*)d_in[13];
    const float* gb2     = (const float*)d_in[14];
    const float* gln2g   = (const float*)d_in[15];
    const float* gln2b   = (const float*)d_in[16];
    const float* deW     = (const float*)d_in[17];
    const float* deb     = (const float*)d_in[18];
    const float* tWqkv   = (const float*)d_in[19];
    const float* tbqkv   = (const float*)d_in[20];
    const float* tWo     = (const float*)d_in[21];
    const float* tbo     = (const float*)d_in[22];
    const float* tln1g   = (const float*)d_in[23];
    const float* tln1b   = (const float*)d_in[24];
    const float* tW1     = (const float*)d_in[25];
    const float* tb1     = (const float*)d_in[26];
    const float* tW2     = (const float*)d_in[27];
    const float* tb2     = (const float*)d_in[28];
    const float* tln2g   = (const float*)d_in[29];
    const float* tln2b   = (const float*)d_in[30];
    const float* W_out   = (const float*)d_in[31];

    // workspace carve-up (floats)
    float* ws   = (float*)d_ws;
    float* P    = ws;                               // NTOK*32
    float* bbuf = P    + (size_t)NTOK * CC;         // NPAIR*8
    float* yA   = bbuf + (size_t)NPAIR * NH;        // NTOK*128
    float* yB   = yA   + (size_t)NTOK * DD;         // NTOK*128
    float* y1   = yB   + (size_t)NTOK * DD;         // NTOK*128
    float* qkvb = y1   + (size_t)NTOK * DD;         // NTOK*384
    float* ao   = qkvb + (size_t)NTOK * 3 * DD;     // NTOK*128
    float* tmp  = ao   + (size_t)NTOK * DD;         // NTOK*128
    float* hid  = tmp  + (size_t)NTOK * DD;         // NTOK*256
    float* pe   = hid  + (size_t)NTOK * HD;         // 62*128

    const int GX = NTOK / 32;   // 589

    k_transpose<<<(NTOK * CC + 255) / 256, 256, 0, stream>>>(pose, P);
    k_pe<<<(TT * DD + 255) / 256, 256, 0, stream>>>(pe);

    // y0 = P @ W_pose^T
    k_gemm<0, false, false><<<dim3(GX, 2), 256, 0, stream>>>(
        P, W_pose, nullptr, nullptr, yA, NTOK, DD, CC);

    float* cur = yA;
    float* oth = yB;
    for (int i = 0; i < 2; i++) {
        // pairwise bias MLP (fused, WMMA)
        k_biasmlp<<<(NPAIR / 16 + 3) / 4, 128, 0, stream>>>(
            P, W_emb + (size_t)i * HD * CC, prelu_a + i,
            W_bias + (size_t)i * NH * HD, bbuf, NPAIR / 16);

        // ---- graph encoder (chunks of V=19, with bias) ----
        k_gemm<0, true, false><<<dim3(GX, 6), 256, 0, stream>>>(
            cur, gWqkv + (size_t)i * 3 * DD * DD, gbqkv + (size_t)i * 3 * DD,
            nullptr, qkvb, NTOK, 3 * DD, DD);
        k_attn<VV, true><<<dim3(NTOK / VV, NH), 64, 0, stream>>>(qkvb, bbuf, ao);
        k_gemm<0, true, false><<<dim3(GX, 2), 256, 0, stream>>>(
            ao, gWo + (size_t)i * DD * DD, gbo + (size_t)i * DD,
            nullptr, tmp, NTOK, DD, DD);
        k_ln<<<NTOK / 8, 256, 0, stream>>>(
            cur, tmp, gln1g + (size_t)i * DD, gln1b + (size_t)i * DD, nullptr, y1, NTOK);
        k_gemm<1, true, false><<<dim3(GX, 4), 256, 0, stream>>>(
            y1, gW1 + (size_t)i * HD * DD, gb1 + (size_t)i * HD,
            nullptr, hid, NTOK, HD, DD);
        k_gemm<0, true, false><<<dim3(GX, 2), 256, 0, stream>>>(
            hid, gW2 + (size_t)i * DD * HD, gb2 + (size_t)i * DD,
            nullptr, tmp, NTOK, DD, HD);
        k_ln<<<NTOK / 8, 256, 0, stream>>>(
            y1, tmp, gln2g + (size_t)i * DD, gln2b + (size_t)i * DD, nullptr, y1, NTOK);
        // de linear + positional encoding -> mem (stored in cur's buffer)
        k_gemm<0, true, true><<<dim3(GX, 2), 256, 0, stream>>>(
            y1, deW + (size_t)i * DD * DD, deb + (size_t)i * DD, pe, cur, NTOK, DD, DD);

        // ---- temporal encoder (chunks of T=62, no bias), mem = cur ----
        k_gemm<0, true, false><<<dim3(GX, 6), 256, 0, stream>>>(
            cur, tWqkv + (size_t)i * 3 * DD * DD, tbqkv + (size_t)i * 3 * DD,
            nullptr, qkvb, NTOK, 3 * DD, DD);
        k_attn<TT, false><<<dim3(NTOK / TT, NH), 64, 0, stream>>>(qkvb, nullptr, ao);
        k_gemm<0, true, false><<<dim3(GX, 2), 256, 0, stream>>>(
            ao, tWo + (size_t)i * DD * DD, tbo + (size_t)i * DD,
            nullptr, tmp, NTOK, DD, DD);
        k_ln<<<NTOK / 8, 256, 0, stream>>>(
            cur, tmp, tln1g + (size_t)i * DD, tln1b + (size_t)i * DD, nullptr, y1, NTOK);
        k_gemm<1, true, false><<<dim3(GX, 4), 256, 0, stream>>>(
            y1, tW1 + (size_t)i * HD * DD, tb1 + (size_t)i * HD,
            nullptr, hid, NTOK, HD, DD);
        k_gemm<0, true, false><<<dim3(GX, 2), 256, 0, stream>>>(
            hid, tW2 + (size_t)i * DD * HD, tb2 + (size_t)i * DD,
            nullptr, tmp, NTOK, DD, HD);
        // final LN + residual (+mem) -> other buffer
        k_ln<<<NTOK / 8, 256, 0, stream>>>(
            y1, tmp, tln2g + (size_t)i * DD, tln2b + (size_t)i * DD, cur, oth, NTOK);

        float* t2 = cur; cur = oth; oth = t2;
    }

    // out = leaky_relu(y @ W_out^T)  -> d_out (flat == reshape(B,-1,T,V))
    k_gemm<2, false, false><<<dim3(GX, 1), 256, 0, stream>>>(
        cur, W_out, nullptr, nullptr, (float*)d_out, NTOK, 64, DD);
}
